// FranchiseSequenceModel_91164975825295
// MI455X (gfx1250) — matrix operand order repaired and checked by reading, not compiled, not run on hardware
//
#include <hip/hip_runtime.h>
#include <hip/hip_bf16.h>

#define B_   128
#define S_   50
#define E_   512
#define D_   1024
#define H_   16
#define L_   4
#define DK_  64
#define FF_  4096
#define V_   100000
#define NR_  5

#define CEILDIV(a,b) (((a)+(b)-1)/(b))

typedef _Float16 f16;
typedef __attribute__((ext_vector_type(16))) _Float16 v16h;
typedef __attribute__((ext_vector_type(8)))  _Float16 v8h;
typedef __attribute__((ext_vector_type(8)))  float    v8f;

union V16 { v16h v; v8h h[2]; };

__device__ __forceinline__ float geluf(float x) {
  return 0.5f * x * (1.0f + erff(x * 0.70710678118654752f));
}

// ---------------------------------------------------------------------------
// WMMA GEMM: C[M,N] = act(A[M,K] @ W[K,N] + bias[N])
// block = 256 threads = 8 waves; block tile 128(M) x 128(N);
// wave tile 32x64 = 2x4 fragments -> 8 v_wmma per K=32 stage.
// Double-buffered LDS + software pipeline: batched global_load_b128 of the
// next stage are issued right after the barrier and overlap the WMMA work of
// the current stage. B is staged transposed so fragment reads are contiguous
// ds_load_b128. M,K multiples of 32 everywhere; N ragged only for V=100000
// (clamped loads / guarded stores; no divergent exits since we use barriers).
// ---------------------------------------------------------------------------
#define APITCH 40   // f16 per A row in LDS (80B pitch: 16B-aligned b128 reads)
#define BPITCH 40   // f16 per B column in LDS

__global__ void __launch_bounds__(256) wmma_gemm(
    const float* __restrict__ A, int lda,
    const float* __restrict__ W, int ldw,
    const float* __restrict__ bias,
    float* __restrict__ C, int ldc,
    int M, int N, int K, int act)
{
  __shared__ f16 As[2][128 * APITCH];   // [row][k]
  __shared__ f16 Bs[2][128 * BPITCH];   // [col][k] (transposed)

  const int tid  = threadIdx.x;
  const int lane = tid & 31;
  const int wave = tid >> 5;
  const int wm   = wave & 3;   // 0..3 -> M sub-tile
  const int wn   = wave >> 2;  // 0..1 -> N sub-tile
  const int m_blk = blockIdx.y * 128;
  const int n_blk = blockIdx.x * 128;

  v8f acc[2][4];
  #pragma unroll
  for (int i = 0; i < 2; ++i)
    #pragma unroll
    for (int j = 0; j < 4; ++j) acc[i][j] = (v8f){};

  // fragment geometry (ISA 16-bit A 16x32 / B 32x16 layouts)
  const int ar0 = wm * 32 + (lane & 15);      // A row for frag 0 (frag 1: +16)
  const int akh = (lane >> 4) * 8;            // A K sub-offset
  const int bc0 = wn * 64 + (lane & 15);      // B col for frag j: +16*j
  const int bkh = (lane >> 4) * 8;            // B K offset

  // staging geometry: A = 128x32 as 4x(32 rows x 8 quads); B = 32x128 as
  // 4x(8 k-rows x 32 quads), stored transposed.
  const int sar = tid >> 3;            // A row base, rows sar+32*i
  const int sac = (tid & 7) * 4;       // A col (k offset)
  const int sbr = tid >> 5;            // B k-row base, rows sbr+8*i
  const int sbc = (tid & 31) * 4;      // B col base
  int nidx = n_blk + sbc;
  if (nidx + 4 > N) nidx = N - 4;      // N multiple of 4 at all call sites
  const float* aPtr = A + (size_t)(m_blk + sar) * lda + sac;
  const float* bPtr = W + (size_t)sbr * ldw + nidx;

  // prologue: batched loads of stage 0
  float4 areg[4], breg[4];
  #pragma unroll
  for (int i = 0; i < 4; ++i) areg[i] = *(const float4*)(aPtr + (size_t)(32 * i) * lda);
  #pragma unroll
  for (int i = 0; i < 4; ++i) breg[i] = *(const float4*)(bPtr + (size_t)(8 * i) * ldw);

  int buf = 0;
  for (int k0 = 0; k0 < K; k0 += 32) {
    // ---- convert + store staged registers into LDS[buf] ----
    #pragma unroll
    for (int i = 0; i < 4; ++i) {
      f16* dst = &As[buf][(sar + 32 * i) * APITCH + sac];
      dst[0] = (f16)areg[i].x; dst[1] = (f16)areg[i].y;
      dst[2] = (f16)areg[i].z; dst[3] = (f16)areg[i].w;
    }
    #pragma unroll
    for (int i = 0; i < 4; ++i) {
      f16* dst = &Bs[buf][sbc * BPITCH + sbr + 8 * i];
      dst[0 * BPITCH] = (f16)breg[i].x;
      dst[1 * BPITCH] = (f16)breg[i].y;
      dst[2 * BPITCH] = (f16)breg[i].z;
      dst[3 * BPITCH] = (f16)breg[i].w;
    }
    __syncthreads();

    // ---- issue next-stage global loads; they overlap the WMMAs below ----
    if (k0 + 32 < K) {
      aPtr += 32;
      bPtr += (size_t)32 * ldw;
      #pragma unroll
      for (int i = 0; i < 4; ++i) areg[i] = *(const float4*)(aPtr + (size_t)(32 * i) * lda);
      #pragma unroll
      for (int i = 0; i < 4; ++i) breg[i] = *(const float4*)(bPtr + (size_t)(8 * i) * ldw);
      if (k0 + 64 < K) {
        __builtin_prefetch(aPtr + 32, 0, 1);
        __builtin_prefetch(bPtr + (size_t)32 * ldw, 0, 1);
      }
    }

    // ---- fragments from LDS[buf] + 8 WMMAs ----
    V16 a0, a1;
    a0.h[0] = *(const v8h*)&As[buf][ar0 * APITCH + akh];
    a0.h[1] = *(const v8h*)&As[buf][ar0 * APITCH + 16 + akh];
    a1.h[0] = *(const v8h*)&As[buf][(ar0 + 16) * APITCH + akh];
    a1.h[1] = *(const v8h*)&As[buf][(ar0 + 16) * APITCH + 16 + akh];
    V16 bf[4];
    #pragma unroll
    for (int j = 0; j < 4; ++j) {
      bf[j].h[0] = *(const v8h*)&Bs[buf][(bc0 + 16 * j) * BPITCH + bkh];
      bf[j].h[1] = *(const v8h*)&Bs[buf][(bc0 + 16 * j) * BPITCH + bkh + 16];
    }
    #pragma unroll
    for (int j = 0; j < 4; ++j) {
      acc[0][j] = __builtin_amdgcn_wmma_f32_16x16x32_f16(false, a0.v, false, bf[j].v, (short)0, acc[0][j], false, false);
      acc[1][j] = __builtin_amdgcn_wmma_f32_16x16x32_f16(false, a1.v, false, bf[j].v, (short)0, acc[1][j], false, false);
    }
    buf ^= 1;
  }

  // ---- epilogue: bias (+gelu) + store ----
  const int mh = (lane >> 4) * 8;
  #pragma unroll
  for (int j = 0; j < 4; ++j) {
    int col = n_blk + wn * 64 + 16 * j + (lane & 15);
    if (col < N) {
      float bvv = bias ? bias[col] : 0.0f;
      #pragma unroll
      for (int i = 0; i < 2; ++i) {
        #pragma unroll
        for (int r = 0; r < 8; ++r) {
          int row = m_blk + wm * 32 + 16 * i + mh + r;
          if (row < M) {
            float t = acc[i][j][r] + bvv;
            if (act) t = geluf(t);
            C[(size_t)row * ldc + col] = t;
          }
        }
      }
    }
  }
}

// ---------------------------------------------------------------------------
// Self-attention fused scores + relative bias + softmax.
// One wave per (b,h,i) row; lanes cover j and j+32 (S=50).
// ---------------------------------------------------------------------------
__global__ void __launch_bounds__(128) attn_scores_softmax(
    const float* __restrict__ q, const float* __restrict__ k,
    const float* __restrict__ rel, const int* __restrict__ relmat,
    float* __restrict__ attn)
{
  int wid = blockIdx.x * 4 + (threadIdx.x >> 5);
  if (wid >= B_ * H_ * S_) return;
  int i = wid % S_;
  int h = (wid / S_) % H_;
  int b = wid / (S_ * H_);
  int lane = threadIdx.x & 31;
  const float* qrow = q + (size_t)(b * S_ + i) * D_ + h * DK_;

  float sc0 = -1e30f, sc1 = -1e30f;
  #pragma unroll
  for (int jj = 0; jj < 2; ++jj) {
    int j = lane + 32 * jj;
    if (j < S_) {
      const float* krow = k + (size_t)(b * S_ + j) * D_ + h * DK_;
      int rid = relmat[i * S_ + j];
      const float* rrow = rel + (size_t)rid * DK_;
      float s1 = 0.f, s2 = 0.f;
      for (int t = 0; t < DK_; ++t) { float qt = qrow[t]; s1 += qt * krow[t]; s2 += qt * rrow[t]; }
      float s = 0.125f * s1 + 0.5f * s2;
      if (jj == 0) sc0 = s; else sc1 = s;
    }
  }
  float m = fmaxf(sc0, sc1);
  for (int off = 16; off; off >>= 1) m = fmaxf(m, __shfl_xor(m, off, 32));
  float e0 = (lane < S_) ? __expf(sc0 - m) : 0.f;
  float e1 = (lane + 32 < S_) ? __expf(sc1 - m) : 0.f;
  float sum = e0 + e1;
  for (int off = 16; off; off >>= 1) sum += __shfl_xor(sum, off, 32);
  float inv = 1.0f / sum;
  if (lane < S_)       attn[(size_t)wid * S_ + lane]      = e0 * inv;
  if (lane + 32 < S_)  attn[(size_t)wid * S_ + lane + 32] = e1 * inv;
}

// ctx[b,i,d] = sum_j attn[b,h,i,j] * v[b,j,d]   (h = d/64)
__global__ void __launch_bounds__(256) ctx_kernel(
    const float* __restrict__ attn, const float* __restrict__ v, float* __restrict__ ctx)
{
  size_t idx = (size_t)blockIdx.x * 256 + threadIdx.x;
  if (idx >= (size_t)B_ * S_ * D_) return;
  int d = (int)(idx % D_);
  int i = (int)((idx / D_) % S_);
  int b = (int)(idx / ((size_t)D_ * S_));
  int h = d >> 6;
  const float* arow = attn + (size_t)((b * H_ + h) * S_ + i) * S_;
  const float* vcol = v + (size_t)b * S_ * D_ + d;
  float s = 0.f;
  for (int j = 0; j < S_; ++j) s += arow[j] * vcol[(size_t)j * D_];
  ctx[idx] = s;
}

// out = LN(a + res) * g + beta ; one 256-thread block per row of D=1024
__global__ void __launch_bounds__(256) ln_residual(
    const float* __restrict__ a, const float* __restrict__ res,
    const float* __restrict__ g, const float* __restrict__ beta,
    float* __restrict__ out)
{
  int row = blockIdx.x;
  const float* pa = a + (size_t)row * D_;
  const float* pr = res + (size_t)row * D_;
  __shared__ float red1[8], red2[8];
  float vals[4];
  float s = 0.f;
  #pragma unroll
  for (int u = 0; u < 4; ++u) { int d = threadIdx.x + u * 256; vals[u] = pa[d] + pr[d]; s += vals[u]; }
  for (int off = 16; off; off >>= 1) s += __shfl_xor(s, off, 32);
  int lane = threadIdx.x & 31, w = threadIdx.x >> 5;
  if (lane == 0) red1[w] = s;
  __syncthreads();
  float mean = 0.f;
  for (int t = 0; t < 8; ++t) mean += red1[t];
  mean *= (1.0f / D_);
  float vs = 0.f;
  #pragma unroll
  for (int u = 0; u < 4; ++u) { float dd = vals[u] - mean; vs += dd * dd; }
  for (int off = 16; off; off >>= 1) vs += __shfl_xor(vs, off, 32);
  if (lane == 0) red2[w] = vs;
  __syncthreads();
  float var = 0.f;
  for (int t = 0; t < 8; ++t) var += red2[t];
  var *= (1.0f / D_);
  float inv = rsqrtf(var + 1e-5f);
  #pragma unroll
  for (int u = 0; u < 4; ++u) {
    int d = threadIdx.x + u * 256;
    out[(size_t)row * D_ + d] = (vals[u] - mean) * inv * g[d] + beta[d];
  }
}

// concat(movie, franchise, entry) embeddings -> xin [B*S, 1536]
__global__ void __launch_bounds__(256) build_xin(
    const float* __restrict__ mt, const float* __restrict__ ft, const float* __restrict__ et,
    const int* __restrict__ mid, const int* __restrict__ fid, const int* __restrict__ eid,
    float* __restrict__ xin)
{
  size_t idx = (size_t)blockIdx.x * 256 + threadIdx.x;
  if (idx >= (size_t)B_ * S_ * (3 * E_)) return;
  int c  = (int)(idx % (3 * E_));
  int rs = (int)(idx / (3 * E_));
  float v;
  if (c < E_)            v = mt[(size_t)mid[rs] * E_ + c];
  else if (c < 2 * E_)   v = ft[(size_t)fid[rs] * E_ + (c - E_)];
  else                   v = et[(size_t)eid[rs] * E_ + (c - 2 * E_)];
  xin[idx] = v;
}

// x += FranchisePositionalEncoding (computed on the fly)
__global__ void __launch_bounds__(256) pe_add(float* __restrict__ x)
{
  size_t idx = (size_t)blockIdx.x * 256 + threadIdx.x;
  if (idx >= (size_t)B_ * S_ * D_) return;
  int d = (int)(idx % D_);
  int spos = (int)((idx / D_) % S_);
  int dd = (d < 512) ? d : d - 512;
  float mult = (d < 512) ? 1.0f : 1.5f;
  int i = dd >> 1;
  float div = __expf((float)(2 * i) * (-9.210340371976184f / 512.0f)); // -ln(1e4)/half
  float arg = (float)spos * div * mult;
  x[idx] += (dd & 1) ? __cosf(arg) : __sinf(arg);
}

// ue[b,:] = user_table[user_ids[b]]
__global__ void __launch_bounds__(256) gather_user(
    const float* __restrict__ ut, const int* __restrict__ uid, float* __restrict__ ue)
{
  int idx = blockIdx.x * 256 + threadIdx.x;
  if (idx >= B_ * D_) return;
  int d = idx % D_, b = idx / D_;
  ue[idx] = ut[(size_t)uid[b] * D_ + d];
}

// cross-attn softmax: one wave per (b,h); a[b,h,s]
__global__ void __launch_bounds__(128) cross_attn_softmax(
    const float* __restrict__ qh, const float* __restrict__ kh, float* __restrict__ a)
{
  int wid = blockIdx.x * 4 + (threadIdx.x >> 5);
  if (wid >= B_ * H_) return;
  int h = wid % H_, b = wid / H_;
  int lane = threadIdx.x & 31;
  const float* qrow = qh + (size_t)b * D_ + h * DK_;
  float sc0 = -1e30f, sc1 = -1e30f;
  #pragma unroll
  for (int jj = 0; jj < 2; ++jj) {
    int j = lane + 32 * jj;
    if (j < S_) {
      const float* krow = kh + (size_t)(b * S_ + j) * D_ + h * DK_;
      float s1 = 0.f;
      for (int t = 0; t < DK_; ++t) s1 += qrow[t] * krow[t];
      float s = 0.125f * s1;
      if (jj == 0) sc0 = s; else sc1 = s;
    }
  }
  float m = fmaxf(sc0, sc1);
  for (int off = 16; off; off >>= 1) m = fmaxf(m, __shfl_xor(m, off, 32));
  float e0 = (lane < S_) ? __expf(sc0 - m) : 0.f;
  float e1 = (lane + 32 < S_) ? __expf(sc1 - m) : 0.f;
  float sum = e0 + e1;
  for (int off = 16; off; off >>= 1) sum += __shfl_xor(sum, off, 32);
  float inv = 1.0f / sum;
  if (lane < S_)      a[(size_t)wid * S_ + lane]      = e0 * inv;
  if (lane + 32 < S_) a[(size_t)wid * S_ + lane + 32] = e1 * inv;
}

// uc[b,d] = sum_s a[b,h,s] * vh[b,s,d]
__global__ void __launch_bounds__(256) uctx_kernel(
    const float* __restrict__ a, const float* __restrict__ vh, float* __restrict__ uc)
{
  int idx = blockIdx.x * 256 + threadIdx.x;
  if (idx >= B_ * D_) return;
  int d = idx % D_, b = idx / D_, h = d >> 6;
  const float* arow = a + (size_t)(b * H_ + h) * S_;
  const float* vcol = vh + (size_t)b * S_ * D_ + d;
  float s = 0.f;
  for (int j = 0; j < S_; ++j) s += arow[j] * vcol[(size_t)j * D_];
  uc[idx] = s;
}

// ci[b, 1024 + d] = mean_s x[b,s,d]
__global__ void __launch_bounds__(256) seqmean_kernel(
    const float* __restrict__ x, float* __restrict__ ci)
{
  int idx = blockIdx.x * 256 + threadIdx.x;
  if (idx >= B_ * D_) return;
  int d = idx % D_, b = idx / D_;
  const float* p = x + (size_t)b * S_ * D_ + d;
  float s = 0.f;
  for (int j = 0; j < S_; ++j) s += p[(size_t)j * D_];
  ci[(size_t)b * (2 * D_) + D_ + d] = s * (1.0f / S_);
}

__global__ void __launch_bounds__(256) copy_cols(
    const float* __restrict__ src, int sld, float* __restrict__ dst, int dld, int rows, int cols)
{
  int idx = blockIdx.x * 256 + threadIdx.x;
  if (idx >= rows * cols) return;
  int c = idx % cols, r = idx / cols;
  dst[(size_t)r * dld + c] = src[(size_t)r * sld + c];
}

// out[b] = act(dot(hin[b,:K], w[:K]) + bias0); one wave per b
__global__ void __launch_bounds__(128) head_reduce(
    const float* __restrict__ hin, const float* __restrict__ w, const float* __restrict__ bias,
    float* __restrict__ out, int K, int do_sigmoid)
{
  int b = blockIdx.x * 4 + (threadIdx.x >> 5);
  if (b >= B_) return;
  int lane = threadIdx.x & 31;
  float s = 0.f;
  for (int t = lane; t < K; t += 32) s += hin[(size_t)b * K + t] * w[t];
  for (int off = 16; off; off >>= 1) s += __shfl_xor(s, off, 32);
  if (lane == 0) {
    float r = s + bias[0];
    if (do_sigmoid) r = 1.0f / (1.0f + __expf(-r));
    out[b] = r;
  }
}

// order: softmax over 2 logits of ho[b,:512] @ o_w2 + o_b2
__global__ void __launch_bounds__(128) order_head(
    const float* __restrict__ ho, const float* __restrict__ w2, const float* __restrict__ b2,
    float* __restrict__ out)
{
  int b = blockIdx.x * 128 + threadIdx.x;
  if (b >= B_) return;
  float s0 = b2[0], s1 = b2[1];
  for (int t = 0; t < 512; ++t) {
    float hv = ho[(size_t)b * 512 + t];
    s0 += hv * w2[t * 2 + 0];
    s1 += hv * w2[t * 2 + 1];
  }
  float m = fmaxf(s0, s1);
  float e0 = __expf(s0 - m), e1 = __expf(s1 - m);
  float inv = 1.0f / (e0 + e1);
  out[b * 2 + 0] = e0 * inv;
  out[b * 2 + 1] = e1 * inv;
}

// ---------------------------------------------------------------------------
static void gemm(const float* A, int lda, const float* W, int ldw, const float* bias,
                 float* C, int ldc, int M, int N, int K, int act, hipStream_t s)
{
  dim3 grid(CEILDIV(N, 128), CEILDIV(M, 128));
  wmma_gemm<<<grid, 256, 0, s>>>(A, lda, W, ldw, bias, C, ldc, M, N, K, act);
}

extern "C" void kernel_launch(void* const* d_in, const int* in_sizes, int n_in,
                              void* d_out, int out_size, void* d_ws, size_t ws_size,
                              hipStream_t stream)
{
  const float* movie_table = (const float*)d_in[0];
  const float* fran_table  = (const float*)d_in[1];
  const float* entry_table = (const float*)d_in[2];
  const float* proj_w = (const float*)d_in[3];
  const float* proj_b = (const float*)d_in[4];
  const float* wq = (const float*)d_in[5];
  const float* bq = (const float*)d_in[6];
  const float* wk = (const float*)d_in[7];
  const float* bk = (const float*)d_in[8];
  const float* wv = (const float*)d_in[9];
  const float* bv = (const float*)d_in[10];
  const float* wo = (const float*)d_in[11];
  const float* bo = (const float*)d_in[12];
  const float* rel_table = (const float*)d_in[13];
  const float* aln_g = (const float*)d_in[14];
  const float* aln_b = (const float*)d_in[15];
  const float* f_w1 = (const float*)d_in[16];
  const float* f_b1 = (const float*)d_in[17];
  const float* f_w2 = (const float*)d_in[18];
  const float* f_b2 = (const float*)d_in[19];
  const float* ln_g = (const float*)d_in[20];
  const float* ln_b = (const float*)d_in[21];
  const float* user_table = (const float*)d_in[22];
  const float* mq_w = (const float*)d_in[23];
  const float* mq_b = (const float*)d_in[24];
  const float* mk_w = (const float*)d_in[25];
  const float* mk_b = (const float*)d_in[26];
  const float* mv_w = (const float*)d_in[27];
  const float* mv_b = (const float*)d_in[28];
  const float* mo_w = (const float*)d_in[29];
  const float* mo_b = (const float*)d_in[30];
  const float* c_w1 = (const float*)d_in[31];
  const float* c_b1 = (const float*)d_in[32];
  const float* c_w2 = (const float*)d_in[33];
  const float* c_b2 = (const float*)d_in[34];
  const float* n_w1 = (const float*)d_in[35];
  const float* n_b1 = (const float*)d_in[36];
  const float* n_w2 = (const float*)d_in[37];
  const float* n_b2 = (const float*)d_in[38];
  const float* r_w1 = (const float*)d_in[39];
  const float* r_b1 = (const float*)d_in[40];
  const float* r_w2 = (const float*)d_in[41];
  const float* r_b2 = (const float*)d_in[42];
  const float* o_w1 = (const float*)d_in[43];
  const float* o_b1 = (const float*)d_in[44];
  const float* o_w2 = (const float*)d_in[45];
  const float* o_b2 = (const float*)d_in[46];
  const int* user_ids      = (const int*)d_in[47];
  const int* movie_ids     = (const int*)d_in[48];
  const int* franchise_ids = (const int*)d_in[49];
  const int* entry_types   = (const int*)d_in[50];
  const int* relation_mat  = (const int*)d_in[51];

  const size_t BS  = (size_t)B_ * S_;
  const size_t BSD = BS * D_;

  // output layout: x | completion | next_logits | rating | order | user_pref
  float* out_x      = (float*)d_out;
  float* out_comp   = out_x + BSD;                       // +128
  float* out_logits = out_comp + B_;                     // +12.8M
  float* out_rating = out_logits + (size_t)B_ * V_;      // +128
  float* out_order  = out_rating + B_;                   // +256
  float* out_upref  = out_order + 2 * B_;                // +131072

  // workspace layout
  float* ws = (float*)d_ws;
  size_t off = 0;
  auto alloc = [&](size_t n) { float* p = ws + off; off += n; return p; };
  float* xin   = alloc(BS * FF_);      // 26.2M floats; aliases ffmid (xin only used once)
  float* ffmid = xin;
  float* xbuf  = alloc(BSD);
  float* qb    = alloc(BSD);
  float* kb    = alloc(BSD);
  float* vb    = alloc(BSD);
  float* scores = alloc((size_t)B_ * H_ * S_ * S_);
  float* ctx   = alloc(BSD);
  float* tmp   = alloc(BSD);
  float* ue    = alloc((size_t)B_ * D_);
  float* qh    = alloc((size_t)B_ * D_);
  float* uctx  = alloc((size_t)B_ * D_);
  float* ci    = alloc((size_t)B_ * 2 * D_);
  float* ri    = alloc((size_t)B_ * 3 * D_);
  float* hc    = alloc((size_t)B_ * D_);
  float* hn    = alloc((size_t)B_ * D_);
  float* hr    = alloc((size_t)B_ * D_);
  float* ho    = alloc((size_t)B_ * (D_ / 2));
  (void)ws_size; (void)in_sizes; (void)n_in; (void)out_size;

  // 1) embeddings -> concat -> proj -> +PE
  build_xin<<<CEILDIV(BS * (3 * E_), 256), 256, 0, stream>>>(
      movie_table, fran_table, entry_table, movie_ids, franchise_ids, entry_types, xin);
  gemm(xin, 3 * E_, proj_w, D_, proj_b, xbuf, D_, (int)BS, D_, 3 * E_, 0, stream);
  pe_add<<<CEILDIV(BSD, 256), 256, 0, stream>>>(xbuf);

  // 2) encoder layers
  for (int l = 0; l < L_; ++l) {
    const size_t wofs = (size_t)l * D_ * D_;
    gemm(xbuf, D_, wq + wofs, D_, bq + (size_t)l * D_, qb, D_, (int)BS, D_, D_, 0, stream);
    gemm(xbuf, D_, wk + wofs, D_, bk + (size_t)l * D_, kb, D_, (int)BS, D_, D_, 0, stream);
    gemm(xbuf, D_, wv + wofs, D_, bv + (size_t)l * D_, vb, D_, (int)BS, D_, D_, 0, stream);
    attn_scores_softmax<<<CEILDIV(B_ * H_ * S_, 4), 128, 0, stream>>>(
        qb, kb, rel_table + (size_t)l * NR_ * DK_, relation_mat, scores);
    ctx_kernel<<<CEILDIV(BSD, 256), 256, 0, stream>>>(scores, vb, ctx);
    gemm(ctx, D_, wo + wofs, D_, bo + (size_t)l * D_, tmp, D_, (int)BS, D_, D_, 0, stream);
    ln_residual<<<(int)BS, 256, 0, stream>>>(tmp, xbuf, aln_g + (size_t)l * D_, aln_b + (size_t)l * D_, xbuf);
    gemm(xbuf, D_, f_w1 + (size_t)l * D_ * FF_, FF_, f_b1 + (size_t)l * FF_, ffmid, FF_, (int)BS, FF_, D_, 1, stream);
    gemm(ffmid, FF_, f_w2 + (size_t)l * FF_ * D_, D_, f_b2 + (size_t)l * D_, tmp, D_, (int)BS, D_, FF_, 0, stream);
    ln_residual<<<(int)BS, 256, 0, stream>>>(tmp, xbuf, ln_g + (size_t)l * D_, ln_b + (size_t)l * D_, xbuf);
  }

  // 3) user cross-attention
  gather_user<<<CEILDIV(B_ * D_, 256), 256, 0, stream>>>(user_table, user_ids, ue);
  gemm(ue, D_, mq_w, D_, mq_b, qh, D_, B_, D_, D_, 0, stream);
  gemm(xbuf, D_, mk_w, D_, mk_b, qb, D_, (int)BS, D_, D_, 0, stream);   // kh -> qb
  gemm(xbuf, D_, mv_w, D_, mv_b, kb, D_, (int)BS, D_, D_, 0, stream);   // vh -> kb
  cross_attn_softmax<<<CEILDIV(B_ * H_, 4), 128, 0, stream>>>(qh, qb, scores);
  uctx_kernel<<<CEILDIV(B_ * D_, 256), 256, 0, stream>>>(scores, kb, uctx);
  // user_pref -> ci[:, :1024] (ldc = 2048)
  gemm(uctx, D_, mo_w, D_, mo_b, ci, 2 * D_, B_, D_, D_, 0, stream);
  // seq_repr -> ci[:, 1024:]
  seqmean_kernel<<<CEILDIV(B_ * D_, 256), 256, 0, stream>>>(xbuf, ci);
  // user_pref output
  copy_cols<<<CEILDIV(B_ * D_, 256), 256, 0, stream>>>(ci, 2 * D_, out_upref, D_, B_, D_);

  // 4) heads
  gemm(ci, 2 * D_, c_w1, D_, c_b1, hc, D_, B_, D_, 2 * D_, 1, stream);
  head_reduce<<<CEILDIV(B_, 4), 128, 0, stream>>>(hc, c_w2, c_b2, out_comp, D_, 1);

  gemm(ci, 2 * D_, n_w1, D_, n_b1, hn, D_, B_, D_, 2 * D_, 1, stream);
  gemm(hn, D_, n_w2, V_, n_b2, out_logits, V_, B_, V_, D_, 0, stream);

  // ri = concat(user_emb, user_pref, seq_repr)
  copy_cols<<<CEILDIV(B_ * D_, 256), 256, 0, stream>>>(ue, D_, ri, 3 * D_, B_, D_);
  copy_cols<<<CEILDIV(B_ * D_, 256), 256, 0, stream>>>(ci, 2 * D_, ri + D_, 3 * D_, B_, D_);
  copy_cols<<<CEILDIV(B_ * D_, 256), 256, 0, stream>>>(ci + D_, 2 * D_, ri + 2 * D_, 3 * D_, B_, D_);
  gemm(ri, 3 * D_, r_w1, D_, r_b1, hr, D_, B_, D_, 3 * D_, 1, stream);
  head_reduce<<<CEILDIV(B_, 4), 128, 0, stream>>>(hr, r_w2, r_b2, out_rating, D_, 0);

  // order: user_pref (ci first half) @ o_w1 -> gelu -> @ o_w2 -> softmax(2)
  gemm(ci, 2 * D_, o_w1, D_ / 2, o_b1, ho, D_ / 2, B_, D_ / 2, D_, 1, stream);
  order_head<<<CEILDIV(B_, 128), 128, 0, stream>>>(ho, o_w2, o_b2, out_order);

  // 5) final x output
  hipMemcpyAsync(out_x, xbuf, BSD * sizeof(float), hipMemcpyDeviceToDevice, stream);
}
